// TrackMamba_30313879175948
// MI455X (gfx1250) — matrix operand order repaired and checked by reading, not compiled, not run on hardware
//
#include <hip/hip_runtime.h>
#include <hip/hip_bf16.h>
#include <math.h>

// ---------------- model constants ----------------
#define BB   2
#define TT   2048
#define DM   512
#define NL   4
#define DI   1024          // 2*DM
#define DS   64
#define HP   64
#define NH   16            // DI/HP
#define DC   4
#define CONV_DIM 1152      // DI + 2*DS
#define DIN  2192          // 2*DI + 2*DS + NH
#define DIN_PAD 2304       // padded to multiple of 128
#define NTR  32
#define MROWS (BB*TT)      // 4096

typedef __attribute__((ext_vector_type(16))) _Float16 v16h;
typedef __attribute__((ext_vector_type(8)))  _Float16 v8h;
typedef __attribute__((ext_vector_type(8)))  float    v8f;

__device__ __forceinline__ float silu_f(float v) { return v / (1.f + __expf(-v)); }

// ---------------- fragment loaders (contiguous k order, same for A and B) ----------------
// A 16x32 f16: lane L holds row m0+L%16; K = k0+(L/16)*16 + h, h=0..15 contiguous
__device__ __forceinline__ v16h load_a_frag(const _Float16* p) {
    v8h lo = *(const v8h*)(p);
    v8h hi = *(const v8h*)(p + 8);
    v16h r;
#pragma unroll
    for (int i = 0; i < 8; ++i) { r[i] = lo[i]; r[i + 8] = hi[i]; }
    return r;
}
// B 32x16 f16: lane L holds column n0+L%16; K = k0+(L/16)*16 + h, h=0..15 contiguous
__device__ __forceinline__ v16h load_b_frag(const _Float16* p) {
    v8h lo = *(const v8h*)(p);
    v8h hi = *(const v8h*)(p + 8);
    v16h r;
#pragma unroll
    for (int i = 0; i < 8; ++i) { r[i] = lo[i]; r[i + 8] = hi[i]; }
    return r;
}

// one wave-tile fragment set: 2 A frags (32 rows) x 4 B frags (64 cols)
struct Frags { v16h a0, a1, b0, b1, b2, b3; };

__device__ __forceinline__ Frags load_set(
    const _Float16* pA0, const _Float16* pA1,
    const _Float16* pB0, const _Float16* pB1,
    const _Float16* pB2, const _Float16* pB3, int k0)
{
    Frags f;
    f.a0 = load_a_frag(pA0 + k0);
    f.a1 = load_a_frag(pA1 + k0);
    f.b0 = load_b_frag(pB0 + k0);
    f.b1 = load_b_frag(pB1 + k0);
    f.b2 = load_b_frag(pB2 + k0);
    f.b3 = load_b_frag(pB3 + k0);
    return f;
}

__device__ __forceinline__ void wmma8(v8f acc[2][4], const Frags& f) {
    acc[0][0] = __builtin_amdgcn_wmma_f32_16x16x32_f16(false, f.a0, false, f.b0, (short)0, acc[0][0], false, false);
    acc[1][0] = __builtin_amdgcn_wmma_f32_16x16x32_f16(false, f.a1, false, f.b0, (short)0, acc[1][0], false, false);
    acc[0][1] = __builtin_amdgcn_wmma_f32_16x16x32_f16(false, f.a0, false, f.b1, (short)0, acc[0][1], false, false);
    acc[1][1] = __builtin_amdgcn_wmma_f32_16x16x32_f16(false, f.a1, false, f.b1, (short)0, acc[1][1], false, false);
    acc[0][2] = __builtin_amdgcn_wmma_f32_16x16x32_f16(false, f.a0, false, f.b2, (short)0, acc[0][2], false, false);
    acc[1][2] = __builtin_amdgcn_wmma_f32_16x16x32_f16(false, f.a1, false, f.b2, (short)0, acc[1][2], false, false);
    acc[0][3] = __builtin_amdgcn_wmma_f32_16x16x32_f16(false, f.a0, false, f.b3, (short)0, acc[0][3], false, false);
    acc[1][3] = __builtin_amdgcn_wmma_f32_16x16x32_f16(false, f.a1, false, f.b3, (short)0, acc[1][3], false, false);
}

// ---------------- WMMA GEMM: C[M,N] = A[M,K] * W[N,K]^T (+bias,+res,act) ---
// block 256 = 8 waves; WG tile 128M x 128N; wave tile 32M x 64N (2x4 wmma tiles).
// Ping-pong software pipeline (manual unroll-by-2, no register copies):
// compute on set S0 while S1's loads are in flight, and vice versa.
// K/32 is even (K in {512,1024,2048}), so pairs divide evenly.
__global__ __launch_bounds__(256) void tm_gemm_wmma(
    const _Float16* __restrict__ A,  int lda,
    const _Float16* __restrict__ W,  int ldb,
    float*          __restrict__ C,  int ldc,
    _Float16*       __restrict__ C16,int ldc16,
    const float*    __restrict__ bias,
    const float*    __restrict__ res, int ldres,
    int K, int act)
{
    const int tid  = threadIdx.x;
    const int lane = tid & 31;
    const int wave = tid >> 5;
    const int wm   = wave & 3;     // 4 wave rows
    const int wn   = wave >> 2;    // 2 wave cols
    const int m0   = blockIdx.y * 128 + wm * 32;
    const int n0   = blockIdx.x * 128 + wn * 64;
    const int l16  = lane & 15;
    const int lhi  = lane >> 4;    // 0/1

    const _Float16* pA0 = A + (size_t)(m0 + l16) * lda + lhi * 16;
    const _Float16* pA1 = pA0 + (size_t)16 * lda;
    const _Float16* pB0 = W + (size_t)(n0 +  0 + l16) * ldb + lhi * 16;
    const _Float16* pB1 = W + (size_t)(n0 + 16 + l16) * ldb + lhi * 16;
    const _Float16* pB2 = W + (size_t)(n0 + 32 + l16) * ldb + lhi * 16;
    const _Float16* pB3 = W + (size_t)(n0 + 48 + l16) * ldb + lhi * 16;

    v8f acc[2][4];
#pragma unroll
    for (int mt = 0; mt < 2; ++mt)
#pragma unroll
        for (int nt = 0; nt < 4; ++nt)
#pragma unroll
            for (int r = 0; r < 8; ++r) acc[mt][nt][r] = 0.f;

    Frags s0 = load_set(pA0, pA1, pB0, pB1, pB2, pB3, 0);
    int k0 = 32;
    const int pairs = (K >> 6) - 1;      // K/64 - 1 full ping-pong pairs
#pragma unroll 1
    for (int i = 0; i < pairs; ++i, k0 += 64) {
        Frags s1 = load_set(pA0, pA1, pB0, pB1, pB2, pB3, k0);
        wmma8(acc, s0);
        s0 = load_set(pA0, pA1, pB0, pB1, pB2, pB3, k0 + 32);
        wmma8(acc, s1);
    }
    // tail pair: k0 == K-32 here
    {
        Frags s1 = load_set(pA0, pA1, pB0, pB1, pB2, pB3, k0);
        wmma8(acc, s0);
        wmma8(acc, s1);
    }

    // epilogue: C/D layout -> row = base + r + lhi*8, col = n0..+l16
#pragma unroll
    for (int mt = 0; mt < 2; ++mt) {
#pragma unroll
        for (int nt = 0; nt < 4; ++nt) {
            const int col = n0 + nt * 16 + l16;
            float bv = bias ? bias[col] : 0.f;
#pragma unroll
            for (int r = 0; r < 8; ++r) {
                const int row = m0 + mt * 16 + lhi * 8 + r;
                float v = acc[mt][nt][r] + bv;
                if (res) v += res[(size_t)row * ldres + col];
                if (act == 1) v = silu_f(v);
                if (C)   C  [(size_t)row * ldc   + col] = v;
                if (C16) C16[(size_t)row * ldc16 + col] = (_Float16)v;
            }
        }
    }
}

// ---------------- weight f32 -> f16 (with N padding) ----------------
__global__ __launch_bounds__(256) void tm_convert_pad(
    const float* __restrict__ src, _Float16* __restrict__ dst,
    int N, int Npad, int K, long total)
{
    long idx = (long)blockIdx.x * 256 + threadIdx.x;
    if (idx >= total) return;
    int  k = (int)(idx % K);
    long nn = idx / K;
    int  n = (int)(nn % Npad);
    long l = nn / Npad;
    float v = (n < N) ? src[(l * (long)N + n) * (long)K + k] : 0.f;
    dst[idx] = (_Float16)v;
}

// ---------------- token embed: one-hot conv(W=3) + bias + pos_emb --------
__global__ __launch_bounds__(256) void tm_embed(
    const int* __restrict__ x, const float* __restrict__ conv_w,
    const float* __restrict__ conv_b, const float* __restrict__ pos,
    float* __restrict__ h, _Float16* __restrict__ h16, int pass)
{
    int idx = blockIdx.x * 256 + threadIdx.x;     // < 4096*512
    int d   = idx & (DM - 1);
    int row = idx >> 9;
    int t   = row & (TT - 1);
    int b   = row >> 11;
    float v = conv_b[d] + pos[(size_t)t * DM + d];
#pragma unroll
    for (int w = 0; w < 3; ++w) {
        int tp = t + w - 1;
        if (tp < 0 || tp >= TT) continue;
        int tok;
        if (pass == 0) tok = x[b * TT + tp];
        else { int raw = x[b * TT + (TT - 1 - tp)]; tok = (raw < 4) ? (3 - raw) : raw; }
        if (tok < 4) v += conv_w[(w * 4 + tok) * DM + d];
    }
    h[idx]   = v;
    h16[idx] = (_Float16)v;
}

// ---------------- causal depthwise conv1d (DC=4) + SiLU -------------------
__global__ __launch_bounds__(256) void tm_conv1d_silu(
    const float* __restrict__ zx,   // bufA, stride DIN_PAD, xBC at cols [DI, DI+CONV_DIM)
    const float* __restrict__ cw, const float* __restrict__ cb,
    float* __restrict__ xbc)        // bufB, stride CONV_DIM
{
    int idx = blockIdx.x * 256 + threadIdx.x;     // < 4096*1152
    int c   = idx % CONV_DIM;
    int row = idx / CONV_DIM;
    int t   = row & (TT - 1);
    float v = cb[c];
#pragma unroll
    for (int w = 0; w < DC; ++w) {
        int tp = t - (DC - 1) + w;
        if (tp < 0) continue;
        v += zx[(size_t)(row - (DC - 1) + w) * DIN_PAD + DI + c] * cw[w * CONV_DIM + c];
    }
    xbc[(size_t)row * CONV_DIM + c] = silu_f(v);
}

// ---------------- selective-state scan: one WG per (b, head) --------------
__global__ __launch_bounds__(256) void tm_scan(
    const float* __restrict__ zx,   // bufA: dt_raw at col 2176+h
    const float* __restrict__ xbc,  // bufB: x | B | C
    float* __restrict__ y,          // bufC, stride DI
    const float* __restrict__ dtb_p, const float* __restrict__ alog_p,
    const float* __restrict__ dssm_p)
{
    const int bh = blockIdx.x;                 // 0..31
    const int b  = bh >> 4;
    const int hI = bh & (NH - 1);
    const int tid = threadIdx.x;
    const int p  = tid >> 2;                   // 0..63
    const int nb = (tid & 3) << 4;             // 0,16,32,48

    float S[16];
#pragma unroll
    for (int j = 0; j < 16; ++j) S[j] = 0.f;

    const float dtb  = dtb_p[hI];
    const float negA = -__expf(alog_p[hI]);
    const float Dh   = dssm_p[hI];

    __shared__ float sX[HP], sB[DS], sC[DS];

    for (int t = 0; t < TT; ++t) {
        const size_t row = (size_t)b * TT + t;
        if (tid < 64)        sX[tid]       = xbc[row * CONV_DIM + hI * HP + tid];
        else if (tid < 128)  sB[tid - 64]  = xbc[row * CONV_DIM + DI + (tid - 64)];
        else if (tid < 192)  sC[tid - 128] = xbc[row * CONV_DIM + DI + DS + (tid - 128)];
        __syncthreads();

        float dtraw = zx[row * DIN_PAD + DI + CONV_DIM + hI] + dtb;
        float dt = (dtraw > 20.f) ? dtraw : log1pf(__expf(dtraw));
        float dA = __expf(negA * dt);
        float xt = sX[p];
        float cf = dt * xt;
        float ys = 0.f;
#pragma unroll
        for (int j = 0; j < 16; ++j) {
            float Sv = S[j] * dA + cf * sB[nb + j];
            S[j] = Sv;
            ys += Sv * sC[nb + j];
        }
        ys += __shfl_xor(ys, 1, 32);
        ys += __shfl_xor(ys, 2, 32);
        if ((tid & 3) == 0)
            y[row * DI + hI * HP + p] = ys + Dh * xt;
        __syncthreads();
    }
}

// ---------------- y * silu(z), RMSNorm, * norm_w, -> f16 ------------------
__global__ __launch_bounds__(256) void tm_gate_rms(
    const float* __restrict__ y, const float* __restrict__ zx,
    const float* __restrict__ nw, _Float16* __restrict__ y16)
{
    const int row = blockIdx.x;
    const int tid = threadIdx.x;
    float vals[4]; float ss = 0.f;
#pragma unroll
    for (int j = 0; j < 4; ++j) {
        int e = tid + j * 256;
        float v = y[(size_t)row * DI + e] * silu_f(zx[(size_t)row * DIN_PAD + e]);
        vals[j] = v; ss += v * v;
    }
    __shared__ float red[256];
    red[tid] = ss; __syncthreads();
    for (int s = 128; s > 0; s >>= 1) {
        if (tid < s) red[tid] += red[tid + s];
        __syncthreads();
    }
    float scale = rsqrtf(red[0] / (float)DI + 1e-5f);
#pragma unroll
    for (int j = 0; j < 4; ++j) {
        int e = tid + j * 256;
        y16[(size_t)row * DI + e] = (_Float16)(vals[j] * scale * nw[e]);
    }
}

// ---------------- h = res + LayerNorm(m); emit f32 + f16 ------------------
__global__ __launch_bounds__(256) void tm_add_ln(
    const float* __restrict__ m, float* __restrict__ h,
    _Float16* __restrict__ h16,
    const float* __restrict__ g, const float* __restrict__ bta)
{
    const int row = blockIdx.x;
    const int tid = threadIdx.x;
    float v0 = m[(size_t)row * DM + tid];
    float v1 = m[(size_t)row * DM + tid + 256];
    __shared__ float r1[256], r2[256];
    r1[tid] = v0 + v1;
    r2[tid] = v0 * v0 + v1 * v1;
    __syncthreads();
    for (int s = 128; s > 0; s >>= 1) {
        if (tid < s) { r1[tid] += r1[tid + s]; r2[tid] += r2[tid + s]; }
        __syncthreads();
    }
    float mean = r1[0] / (float)DM;
    float var  = r2[0] / (float)DM - mean * mean;
    float inv  = rsqrtf(var + 1e-5f);
#pragma unroll
    for (int j = 0; j < 2; ++j) {
        int d = tid + j * 256;
        float v = (j == 0 ? v0 : v1);
        float o = h[(size_t)row * DM + d] + (v - mean) * inv * g[d] + bta[d];
        h[(size_t)row * DM + d]   = o;
        h16[(size_t)row * DM + d] = (_Float16)o;
    }
}

// ---------------- heads: y_r / y_c, averaged over both passes -------------
__global__ __launch_bounds__(256) void tm_heads(
    const float* __restrict__ h,
    const float* __restrict__ rw, const float* __restrict__ rb,
    const float* __restrict__ cw, const float* __restrict__ cb,
    float* __restrict__ out, int pass)
{
    const int row  = blockIdx.x;          // 0..4095
    const int tid  = threadIdx.x;
    const int o    = tid >> 2;            // 0..63
    const int part = tid & 3;
    const int head = o >> 5;
    const int n    = o & 31;
    const float* w  = head ? cw : rw;
    const float* bb = head ? cb : rb;
    const float* hr = h + (size_t)row * DM + part * 128;
    float s = 0.f;
    for (int j = 0; j < 128; ++j)
        s += hr[j] * w[(size_t)(part * 128 + j) * NTR + n];
    s += __shfl_xor(s, 1, 32);
    s += __shfl_xor(s, 2, 32);
    if (part == 0) {
        float v = 0.5f * (s + bb[n]);
        int b = row >> 11, t = row & (TT - 1);
        size_t orow = pass ? (size_t)(b * TT + (TT - 1 - t)) : (size_t)row;
        size_t oidx = (size_t)head * MROWS * NTR + orow * NTR + n;
        if (pass == 0) out[oidx] = v;
        else           out[oidx] += v;
    }
}

// ---------------- host orchestration ----------------
extern "C" void kernel_launch(void* const* d_in, const int* in_sizes, int n_in,
                              void* d_out, int out_size, void* d_ws, size_t ws_size,
                              hipStream_t stream) {
    const int*   x        = (const int*)  d_in[0];
    const float* pos_emb  = (const float*)d_in[1];
    const float* conv_w   = (const float*)d_in[2];
    const float* conv_b   = (const float*)d_in[3];
    const float* in_proj  = (const float*)d_in[4];
    const float* conv1d_w = (const float*)d_in[5];
    const float* conv1d_b = (const float*)d_in[6];
    const float* dt_bias  = (const float*)d_in[7];
    const float* A_log    = (const float*)d_in[8];
    const float* D_ssm    = (const float*)d_in[9];
    const float* norm_w   = (const float*)d_in[10];
    const float* out_w    = (const float*)d_in[11];
    const float* ln_g     = (const float*)d_in[12];
    const float* ln_b     = (const float*)d_in[13];
    const float* mlp_w1   = (const float*)d_in[14];
    const float* mlp_b1   = (const float*)d_in[15];
    const float* mlp_w2   = (const float*)d_in[16];
    const float* mlp_b2   = (const float*)d_in[17];
    const float* reg_w    = (const float*)d_in[18];
    const float* reg_b    = (const float*)d_in[19];
    const float* cls_w    = (const float*)d_in[20];
    const float* cls_b    = (const float*)d_in[21];
    float* out = (float*)d_out;

    char* ws = (char*)d_ws;
    size_t off = 0;
    auto alloc = [&](size_t bytes) -> void* {
        void* p = ws + off;
        off += (bytes + 255) & ~(size_t)255;
        return p;
    };
    float*    h     = (float*)   alloc((size_t)MROWS * DM * 4);
    float*    bufA  = (float*)   alloc((size_t)MROWS * DIN_PAD * 4);
    float*    bufB  = (float*)   alloc((size_t)MROWS * CONV_DIM * 4);
    float*    bufC  = (float*)   alloc((size_t)MROWS * DI * 4);
    float*    bufD  = (float*)   alloc((size_t)MROWS * DM * 4);
    _Float16* h16   = (_Float16*)alloc((size_t)MROWS * DM * 2);
    _Float16* y16   = (_Float16*)alloc((size_t)MROWS * DI * 2);
    _Float16* m16   = (_Float16*)alloc((size_t)MROWS * 2048 * 2);
    _Float16* wIn16 = (_Float16*)alloc((size_t)NL * DIN_PAD * DM * 2);
    _Float16* wOu16 = (_Float16*)alloc((size_t)NL * DM * DI * 2);
    _Float16* wM116 = (_Float16*)alloc((size_t)NL * 2048 * DM * 2);
    _Float16* wM216 = (_Float16*)alloc((size_t)NL * DM * 2048 * 2);

    // weights -> f16 (in_proj padded DIN -> DIN_PAD with zeros)
    {
        long tot;
        tot = (long)NL * DIN_PAD * DM;
        tm_convert_pad<<<(unsigned)((tot + 255) / 256), 256, 0, stream>>>(in_proj, wIn16, DIN, DIN_PAD, DM, tot);
        tot = (long)NL * DM * DI;
        tm_convert_pad<<<(unsigned)((tot + 255) / 256), 256, 0, stream>>>(out_w, wOu16, DM, DM, DI, tot);
        tot = (long)NL * 2048 * DM;
        tm_convert_pad<<<(unsigned)((tot + 255) / 256), 256, 0, stream>>>(mlp_w1, wM116, 2048, 2048, DM, tot);
        tot = (long)NL * DM * 2048;
        tm_convert_pad<<<(unsigned)((tot + 255) / 256), 256, 0, stream>>>(mlp_w2, wM216, DM, DM, 2048, tot);
    }

    for (int pass = 0; pass < 2; ++pass) {
        tm_embed<<<(MROWS * DM) / 256, 256, 0, stream>>>(x, conv_w, conv_b, pos_emb, h, h16, pass);
        for (int l = 0; l < NL; ++l) {
            // zxbcdt = h @ in_proj^T    [4096 x 2192(pad 2304)], K=512
            tm_gemm_wmma<<<dim3(DIN_PAD / 128, MROWS / 128), 256, 0, stream>>>(
                h16, DM, wIn16 + (size_t)l * DIN_PAD * DM, DM,
                bufA, DIN_PAD, nullptr, 0, nullptr, nullptr, 0, DM, 0);
            // causal depthwise conv + silu on xBC
            tm_conv1d_silu<<<(MROWS * CONV_DIM) / 256, 256, 0, stream>>>(
                bufA, conv1d_w + (size_t)l * DC * CONV_DIM, conv1d_b + (size_t)l * CONV_DIM, bufB);
            // sequential selective scan
            tm_scan<<<BB * NH, 256, 0, stream>>>(
                bufA, bufB, bufC, dt_bias + (size_t)l * NH, A_log + (size_t)l * NH, D_ssm + (size_t)l * NH);
            // gate by silu(z), RMSNorm, -> f16
            tm_gate_rms<<<MROWS, 256, 0, stream>>>(bufC, bufA, norm_w + (size_t)l * DI, y16);
            // out projection  [4096 x 512], K=1024
            tm_gemm_wmma<<<dim3(DM / 128, MROWS / 128), 256, 0, stream>>>(
                y16, DI, wOu16 + (size_t)l * DM * DI, DI,
                bufD, DM, nullptr, 0, nullptr, nullptr, 0, DI, 0);
            // h = h + LayerNorm(mamba_out)
            tm_add_ln<<<MROWS, 256, 0, stream>>>(bufD, h, h16, ln_g + (size_t)l * DM, ln_b + (size_t)l * DM);
            // m = silu(h @ W1^T + b1) -> f16 only   [4096 x 2048], K=512
            tm_gemm_wmma<<<dim3(2048 / 128, MROWS / 128), 256, 0, stream>>>(
                h16, DM, wM116 + (size_t)l * 2048 * DM, DM,
                nullptr, 0, m16, 2048, mlp_b1 + (size_t)l * 2048, nullptr, 0, DM, 1);
            // h = h + m @ W2^T + b2  (residual fused)  [4096 x 512], K=2048
            tm_gemm_wmma<<<dim3(DM / 128, MROWS / 128), 256, 0, stream>>>(
                m16, 2048, wM216 + (size_t)l * DM * 2048, 2048,
                h, DM, h16, DM, mlp_b2 + (size_t)l * DM, h, DM, 2048, 0);
        }
        tm_heads<<<MROWS, 256, 0, stream>>>(h, reg_w, reg_b, cls_w, cls_b, out, pass);
    }
    (void)in_sizes; (void)n_in; (void)out_size; (void)ws_size;
}
